// KPConv_62062277427552
// MI455X (gfx1250) — matrix operand order repaired and checked by reading, not compiled, not run on hardware
//
#include <hip/hip_runtime.h>

typedef _Float16 h16 __attribute__((ext_vector_type(16)));
typedef _Float16 h2  __attribute__((ext_vector_type(2)));
typedef float    f8  __attribute__((ext_vector_type(8)));

#define NNB 32
#define CIN 64
#define COUT 64
#define NK 15
#define INV_EXT (1.0f/0.06f)

#define STRIDE_H 1032           // halfs per row (1024 + 8 pad) for W2 and WT

// LDS byte offsets
#define OFF_KP 0                                   // 48 floats of kernel points
#define OFF_W2 192                                 // W2: [64 d][STRIDE_H] f16
#define OFF_WT (OFF_W2 + 64*STRIDE_H*2)            // WT: [4 pairs][16 pts][STRIDE_H] f16
#define OFF_XS (OFF_WT + 4*16*STRIDE_H*2)          // x stage: [8 waves][2 bufs][2048 B]
#define LDS_BYTES (OFF_XS + 8*2*2048)              // 297,152 B <= 320 KB/WGP

// --- CDNA5 async global->LDS copy (ASYNCcnt-tracked) ---------------------
__device__ __forceinline__ void async_copy_b128(unsigned lds_off, const float* g) {
    unsigned long long ga = (unsigned long long)(const void*)g;
    asm volatile("global_load_async_to_lds_b128 %0, %1, off"
                 :: "v"(lds_off), "v"(ga) : "memory");
}
#define WAIT_ASYNC(n) asm volatile("s_wait_asynccnt " #n ::: "memory")

__global__ __launch_bounds__(256)
void kpconv_wmma(const float* __restrict__ q_pts,
                 const float* __restrict__ neighbors,
                 const float* __restrict__ x,
                 const float* __restrict__ kernel_points,
                 const float* __restrict__ weights,
                 float* __restrict__ out,
                 int nPts, int nTiles)
{
    extern __shared__ char lds[];
    float*    kp = (float*)(lds + OFF_KP);
    _Float16* w2 = (_Float16*)(lds + OFF_W2);     // [64][STRIDE_H], idx = c*16 + k
    _Float16* wt = (_Float16*)(lds + OFF_WT);     // [4][16][STRIDE_H]

    const int tid  = threadIdx.x;
    const int lane = tid & 31;
    const int wv   = tid >> 5;
    const int pair = wv >> 1;     // two waves share one 16-point phase-2 GEMM
    const int half = wv & 1;
    const int lid  = lane & 15;
    const int g    = lane >> 4;

    // ---- stage kernel points (once) ----
    if (tid < 48) kp[tid] = (tid < NK * 3) ? kernel_points[tid] : 0.0f;

    // ---- stage weights -> f16 LDS, layout [d][c*16+k] (coalesced global reads) ----
    for (int e = tid; e < NK * CIN * COUT; e += 256) {
        int kk = e / (CIN * COUT);
        int c  = (e / COUT) % CIN;
        int d  = e % COUT;
        w2[d * STRIDE_H + c * 16 + kk] = (_Float16)weights[e];
    }
    for (int e = tid; e < CIN * COUT; e += 256) {   // zero the k==15 pad slots
        int d = e / CIN, c = e % CIN;
        w2[d * STRIDE_H + c * 16 + 15] = (_Float16)0.0f;
    }
    __syncthreads();

    // per-lane kernel point: A-row k == lid (row 15 is the zero pad row)
    const float kx = (lid < NK) ? kp[lid * 3 + 0] : 0.0f;
    const float ky = (lid < NK) ? kp[lid * 3 + 1] : 0.0f;
    const float kz = (lid < NK) ? kp[lid * 3 + 2] : 0.0f;

    _Float16* wtp = wt + pair * (16 * STRIDE_H);
    const unsigned xsBase = (unsigned)(OFF_XS + wv * 4096);   // 2 x 2KB ping-pong

    for (int t = blockIdx.x; t < nTiles; t += gridDim.x) {
        const int base = t * 64;
        __syncthreads();   // WT reuse across grid-stride iterations

        // chunk pc = p*4 + ct : one 32m x 16c f32 slice of x per chunk (2 KB)
        auto issue = [&](int pc) {
            int p  = pc >> 2, ct = pc & 3;
            int n  = base + pair * 16 + half * 8 + p;
            int nc = (n < nPts) ? n : (nPts - 1);
            const float* xg = x + (long)nc * (NNB * CIN) + ct * 16;
            unsigned lb = xsBase + (unsigned)(pc & 1) * 2048u;
#pragma unroll
            for (int r = 0; r < 4; ++r) {
                int e  = lane + r * 32;    // 0..127 transfers of 16 B
                int m  = e >> 2;
                int cq = e & 3;
                async_copy_b128(lb + (unsigned)(m * 64 + cq * 16),
                                xg + m * CIN + cq * 4);
            }
        };
        issue(0);
        issue(1);

        // ================= Phase 1: weighted = w(16x32) @ x(32x64), per point =====
        h16 a;
        for (int pc = 0; pc < 32; ++pc) {
            const int p  = pc >> 2, ct = pc & 3;
            const int pp = half * 8 + p;
            const int n  = base + pair * 16 + pp;
            const int nc = (n < nPts) ? n : (nPts - 1);

            if (ct == 0) {
                // lane holds rel of neighbor m = lane; A-build gathers via shuffles
                float rx = neighbors[nc * (NNB * 3) + lane * 3 + 0] - q_pts[nc * 3 + 0];
                float ry = neighbors[nc * (NNB * 3) + lane * 3 + 1] - q_pts[nc * 3 + 1];
                float rz = neighbors[nc * (NNB * 3) + lane * 3 + 2] - q_pts[nc * 3 + 2];
#pragma unroll
                for (int i = 0; i < 16; ++i) {
                    int m = ((i < 8) ? i : (i + 8)) + g * 8;
                    float dx = __shfl(rx, m, 32) - kx;
                    float dy = __shfl(ry, m, 32) - ky;
                    float dz = __shfl(rz, m, 32) - kz;
                    float d2 = dx * dx + dy * dy + dz * dz;
                    float wgt = 1.0f - sqrtf(d2) * INV_EXT;
                    wgt = (wgt > 0.0f) ? wgt : 0.0f;
                    a[i] = (_Float16)((lid < NK) ? wgt : 0.0f);
                }
            }

            // wait for this chunk's async copies (in-order completion, own wave's cnt)
            if (pc < 31) { WAIT_ASYNC(4); } else { WAIT_ASYNC(0); }

            const float* xs = (const float*)(lds + xsBase + (unsigned)(pc & 1) * 2048u);
            // B operand: element i -> B[m = {0..7,16..23}[i]+8g][c = ct*16+lid]
            h16 b;
#pragma unroll
            for (int i = 0; i < 16; ++i)
                b[i] = (_Float16)xs[(((i < 8) ? i : (i + 8)) + 8 * g) * 16 + lid];

            f8 acc = {0.f, 0.f, 0.f, 0.f, 0.f, 0.f, 0.f, 0.f};
            acc = __builtin_amdgcn_wmma_f32_16x16x32_f16(
                      false, a, false, b, (short)0, acc, false, false);

            // D element j: M=k=2j+8g, N=c=ct*16+lid -> wt[pp][c*16+k] (h2 pairs)
#pragma unroll
            for (int j = 0; j < 4; ++j) {
                h2 v;
                v.x = (_Float16)acc[2 * j];
                v.y = (_Float16)acc[2 * j + 1];
                int kk = 2 * j + 8 * g;
                int c  = ct * 16 + lid;
                *(h2*)(wtp + pp * STRIDE_H + c * 16 + kk) = v;
            }

            if (pc + 2 < 32) issue(pc + 2);   // refill the buffer just consumed
        }
        __syncthreads();

        // ===== Phase 2: out(16x64) = weighted(16x1024) @ W2(1024x64), pair-shared ==
        f8 accs[2];
#pragma unroll
        for (int u = 0; u < 2; ++u)
            accs[u] = (f8){0.f, 0.f, 0.f, 0.f, 0.f, 0.f, 0.f, 0.f};

        for (int s = 0; s < 32; ++s) {
            // A: row = point = lid, K = 32s + {0..7,16..23} + 8g
            h16 av;
#pragma unroll
            for (int j = 0; j < 8; ++j) {
                int i    = 2 * j;
                int kidx = 32 * s + ((i < 8) ? i : (i + 8)) + 8 * g;
                h2 v = *(const h2*)(wtp + lid * STRIDE_H + kidx);
                av[i]     = v.x;
                av[i + 1] = v.y;
            }
#pragma unroll
            for (int u = 0; u < 2; ++u) {
                int dt = half * 2 + u;   // each wave of the pair owns 2 d-tiles
                h16 bv;
#pragma unroll
                for (int j = 0; j < 8; ++j) {
                    int kidx = 32 * s + ((2 * j < 8) ? 2 * j : (2 * j + 8)) + 8 * g;
                    h2 v = *(const h2*)(w2 + (dt * 16 + lid) * STRIDE_H + kidx);
                    bv[2 * j]     = v.x;
                    bv[2 * j + 1] = v.y;
                }
                accs[u] = __builtin_amdgcn_wmma_f32_16x16x32_f16(
                              false, av, false, bv, (short)0, accs[u], false, false);
            }
        }

        // ---- store: D element j -> point p = j+8g, d = dt*16+lid ----
#pragma unroll
        for (int u = 0; u < 2; ++u) {
            int dt = half * 2 + u;
#pragma unroll
            for (int j = 0; j < 8; ++j) {
                int p = j + 8 * g;
                int n = base + pair * 16 + p;
                if (n < nPts) out[n * COUT + dt * 16 + lid] = accs[u][j];
            }
        }
    }
}

extern "C" void kernel_launch(void* const* d_in, const int* in_sizes, int n_in,
                              void* d_out, int out_size, void* d_ws, size_t ws_size,
                              hipStream_t stream) {
    const float* q_pts         = (const float*)d_in[0];
    // d_in[1] (s_pts) and d_in[3] (neighb_inds) are unused by the reference math
    const float* neighbors     = (const float*)d_in[2];
    const float* x             = (const float*)d_in[4];
    const float* kernel_points = (const float*)d_in[5];
    const float* weights       = (const float*)d_in[6];
    float*       out           = (float*)d_out;

    const int nPts   = in_sizes[0] / 3;           // 50000
    const int nTiles = (nPts + 63) / 64;
    int grid = nTiles < 320 ? nTiles : 320;       // grid-stride; amortizes W staging

    kpconv_wmma<<<grid, 256, (size_t)LDS_BYTES, stream>>>(
        q_pts, neighbors, x, kernel_points, weights, out, nPts, nTiles);
}
